// MoE_58042188038317
// MI455X (gfx1250) — matrix operand order repaired and checked by reading, not compiled, not run on hardware
//
#include <hip/hip_runtime.h>

// ---------------- types ----------------
typedef __attribute__((ext_vector_type(16))) _Float16 v16h;
typedef __attribute__((ext_vector_type(8)))  _Float16 v8h;
typedef __attribute__((ext_vector_type(8)))  float    v8f;

union AFrag { v16h v; v8h h8[2]; unsigned int u[8]; _Float16 h[16]; };
union BFrag { v16h v; _Float16 h[16]; };

__device__ __forceinline__ v8f wmma16(v16h a, v16h b, v8f c) {
  // (neg_a, A, neg_b, B, c_mod, C, reuse_a, reuse_b)
  return __builtin_amdgcn_wmma_f32_16x16x32_f16(false, a, false, b, (short)0, c,
                                                false, false);
}

// ---------------- geometry ----------------
#define TB   16          // output tile 16x16
#define P2   20          // input patch extent (halo 2)
#define P1   18          // h / he extent (halo 1)
#define NP2  (P2 * P2)   // 400
#define NP1  (P1 * P1)   // 324
#define XSS  24          // xs row stride in halves (20 ch padded)

// ---------------- LDS carve (halves then floats) ----------------
// Weight B-panels are padded to full WMMA geometry (16 N-rows, K mult of 32,
// zeros in padding) so fragment gathers need no guards at all.
constexpr int XS_H = NP2 * XSS;      // 9600  : x patch, f16, [pixel][24]
constexpr int HB_H = NP1 * 8;        // 2592  : h, f16, [pixel][8]
constexpr int HE_H = NP1 * 64;       // 20736 : he, f16, [pixel][64]
constexpr int PT_H = 9 * 8 * 256;    // 18432 : parts, f16, [(p*8+c)][pixel]
constexpr int W1_H = 16 * 128;       // 2048  : stem conv1 B [16][128], k=tap*12+ch
constexpr int W2_H = 16 * 96;        // 1536  : stem conv2 B [16][96],  k=tap*8+ch
constexpr int EW_H = 8 * 16 * 96;    // 12288 : expert conv2 B [e][16][96]
constexpr int GW_H = 16 * 192;       // 3072  : gate conv1 B [16][192], k=tap*20+ch
constexpr int HALVES = XS_H + HB_H + HE_H + PT_H + W1_H + W2_H + EW_H + GW_H; // 70304
constexpr int GB_F = 256 * 12;       // gate conv1 output f32
constexpr int FLOATS = GB_F + 576 /*dw*/ + 108 /*gw2*/ + 174 /*biases*/;      // 3930
constexpr size_t SMEM_BYTES = (size_t)HALVES * 2 + (size_t)FLOATS * 4;        // 156328

__global__ __launch_bounds__(256)
void moe_fused_kernel(const float* __restrict__ x,
                      const float* __restrict__ w_s1, const float* __restrict__ b_s1,
                      const float* __restrict__ w_s2, const float* __restrict__ b_s2,
                      const float* __restrict__ w_e1, const float* __restrict__ b_e1,
                      const float* __restrict__ w_e2, const float* __restrict__ b_e2,
                      const float* __restrict__ w_g1, const float* __restrict__ b_g1,
                      const float* __restrict__ w_g2, const float* __restrict__ b_g2,
                      const float* __restrict__ w_f,  const float* __restrict__ b_f,
                      float* __restrict__ out) {
  extern __shared__ char smem_raw[];
  _Float16* xs    = (_Float16*)smem_raw;      // [400][24]
  _Float16* hbuf  = xs    + XS_H;             // [324][8]
  _Float16* hebuf = hbuf  + HB_H;             // [324][64]
  _Float16* parts = hebuf + HE_H;             // [9*8][256]
  _Float16* w1B   = parts + PT_H;             // [16][128]
  _Float16* w2B   = w1B   + W1_H;             // [16][96]
  _Float16* ewB   = w2B   + W2_H;             // [8][16][96]
  _Float16* gwB   = ewB   + EW_H;             // [16][192]
  float* gbuf  = (float*)(gwB + GW_H);        // [256][12]
  float* dwW   = gbuf + GB_F;                 // [64][9] depthwise weights
  float* gw2v  = dwW  + 576;                  // [9][12]
  float* s1bv  = gw2v + 108;                  // 8
  float* s2bv  = s1bv + 8;                    // 8
  float* e1bv  = s2bv + 8;                    // 64
  float* e2bv  = e1bv + 64;                   // 64
  float* g1bv  = e2bv + 64;                   // 12
  float* g2bv  = g1bv + 12;                   // 9
  float* fwv   = g2bv + 9;                    // 8
  float* fbv   = fwv  + 8;                    // 1

  const int tid  = threadIdx.x;
  const int lane = tid & 31;
  const int wv   = tid >> 5;
  const int mrow = lane & 15;   // A-row / D-column
  const int hsel = lane >> 4;   // lane half

  const int x0 = blockIdx.x * TB, y0 = blockIdx.y * TB, b = blockIdx.z;
  const float* xb = x + (size_t)b * 20 * 512 * 512;

  // ---------- stage W: weights -> LDS (f16, zero-padded GEMM-B panels) ----------
  for (int i = tid; i < W1_H; i += 256) {               // stem conv1 [16][128]
    int n = i >> 7, k = i & 127;
    float v = 0.f;
    if (n < 8 && k < 108) {
      int tap = k / 12, ch = k - tap * 12;
      v = w_s1[(n * 12 + ch) * 9 + tap];
    }
    w1B[i] = (_Float16)v;
  }
  for (int i = tid; i < W2_H; i += 256) {               // stem conv2 [16][96]
    int n = i / 96, k = i - n * 96;
    float v = 0.f;
    if (n < 8 && k < 72) { int tap = k >> 3, ch = k & 7; v = w_s2[(n * 8 + ch) * 9 + tap]; }
    w2B[i] = (_Float16)v;
  }
  for (int i = tid; i < EW_H; i += 256) {               // expert conv2 [8][16][96]
    int e = i / 1536, rem = i - e * 1536;
    int n = rem / 96, k = rem - n * 96;
    float v = 0.f;
    if (n < 8 && k < 72) { int tap = k >> 3, ch = k & 7; v = w_e2[((e * 8 + n) * 8 + ch) * 9 + tap]; }
    ewB[i] = (_Float16)v;
  }
  for (int i = tid; i < GW_H; i += 256) {               // gate conv1 [16][192]
    int n = i / 192, k = i - n * 192;
    float v = 0.f;
    if (n < 12 && k < 180) {
      int tap = k / 20, ch = k - tap * 20;
      v = w_g1[(n * 20 + ch) * 9 + tap];
    }
    gwB[i] = (_Float16)v;
  }
  for (int i = tid; i < 576; i += 256) dwW[i] = w_e1[i];    // depthwise (64,9)
  if (tid < 108) gw2v[tid] = w_g2[tid];
  if (tid < 8)  { s1bv[tid] = b_s1[tid]; s2bv[tid] = b_s2[tid]; fwv[tid] = w_f[tid]; }
  if (tid < 64) { e1bv[tid] = b_e1[tid]; e2bv[tid] = b_e2[tid]; }
  if (tid < 12) g1bv[tid] = b_g1[tid];
  if (tid < 9)  g2bv[tid] = b_g2[tid];
  if (tid == 0) fbv[0] = b_f[0];

  // ---------- stage A: x patch (halo 2, zero padded) -> LDS f16 ----------
  for (int i = tid; i < 20 * NP2; i += 256) {
    int ch = i / NP2, p = i - ch * NP2;
    int py = p / P2, px = p - py * P2;
    int gy = y0 + py - 2, gx = x0 + px - 2;
    float v = 0.f;
    if (gy >= 0 && gy < 512 && gx >= 0 && gx < 512)
      v = xb[(size_t)ch * 262144 + gy * 512 + gx];
    xs[p * XSS + ch] = (_Float16)v;
  }
  __syncthreads();

  // ---------- stage B: h = relu(conv 12->8) on 18x18, WMMA K=108 ----------
  // A gather: branchless channel-pair (b32) loads; k clamped, B zero-padded.
  for (int t = wv; t < 21; t += 8) {
    v8f acc = {};
    int pix = t * 16 + mrow; if (pix > NP1 - 1) pix = NP1 - 1;
    int py = pix / P1, px = pix - py * P1;
#pragma unroll
    for (int kb = 0; kb < 108; kb += 32) {
      AFrag a;
#pragma unroll
      for (int p = 0; p < 8; ++p) {
        int k = kb + 2 * (p & 3) + 16 * hsel + 8 * (p >> 2);
        int kc = k < 106 ? k : 106;                 // pair start clamp (even -> never crosses tap)
        int tap = kc / 12, ch = kc - tap * 12;
        int ty = tap / 3, tx = tap - ty * 3;
        a.u[p] = *(const unsigned int*)&xs[((py + ty) * P2 + (px + tx)) * XSS + ch];
      }
      BFrag bf;
      bf.v = *(const v16h*)&w1B[mrow * 128 + kb + 16 * hsel];
      acc = wmma16(a.v, bf.v, acc);
    }
    if (mrow < 8) {
      float bias = s1bv[mrow];
#pragma unroll
      for (int r = 0; r < 8; ++r) {
        int op = t * 16 + r + 8 * hsel;
        if (op < NP1) {
          float v = acc[r] + bias;
          hbuf[op * 8 + mrow] = (_Float16)(v > 0.f ? v : 0.f);
        }
      }
    }
  }
  __syncthreads();

  // ---------- stage C: depthwise expert conv he (VALU, K=9) ----------
  for (int i = tid; i < 64 * NP1; i += 256) {
    int c = i & 63, pix = i >> 6;
    int e = c >> 3;
    int py = pix / P1, px = pix - py * P1;
    float acc = e1bv[c];
#pragma unroll
    for (int tap = 0; tap < 9; ++tap) {
      int ty = tap / 3, tx = tap - ty * 3;
      acc += dwW[c * 9 + tap] *
             (float)xs[((py + ty) * P2 + (px + tx)) * XSS + 12 + e];
    }
    hebuf[pix * 64 + c] = (_Float16)(acc > 0.f ? acc : 0.f);
  }

  // ---------- stage P0: p0 = relu(conv 8->8) on 16x16, WMMA, vector LDS ----------
  for (int t = wv; t < 16; t += 8) {
    v8f acc = {};
    int pix = t * 16 + mrow;
    int py = pix >> 4, px = pix & 15;
#pragma unroll
    for (int kb = 0; kb < 96; kb += 32) {
      AFrag a;
#pragma unroll
      for (int r2 = 0; r2 < 2; ++r2) {
        int k0 = kb + 16 * hsel + 8 * r2;
        int tap = k0 >> 3; tap = tap < 8 ? tap : 8;  // clamp: B zero-padded beyond k=72
        int ty = tap / 3, tx = tap - ty * 3;
        a.h8[r2] = *(const v8h*)&hbuf[((py + ty) * P1 + (px + tx)) * 8];
      }
      BFrag bf;
      bf.v = *(const v16h*)&w2B[mrow * 96 + kb + 16 * hsel];
      acc = wmma16(a.v, bf.v, acc);
    }
    if (mrow < 8) {
      float bias = s2bv[mrow];
#pragma unroll
      for (int r = 0; r < 8; ++r) {
        int op = t * 16 + r + 8 * hsel;
        float v = acc[r] + bias;
        parts[(0 * 8 + mrow) * 256 + op] = (_Float16)(v > 0.f ? v : 0.f);
      }
    }
  }

  // ---------- stage G: gate conv1 20->12 on 16x16, WMMA K=180 ----------
  for (int t = wv; t < 16; t += 8) {
    v8f acc = {};
    int pix = t * 16 + mrow;
    int py = pix >> 4, px = pix & 15;
#pragma unroll
    for (int kb = 0; kb < 180; kb += 32) {
      AFrag a;
#pragma unroll
      for (int p = 0; p < 8; ++p) {
        int k = kb + 2 * (p & 3) + 16 * hsel + 8 * (p >> 2);
        int kc = k < 178 ? k : 178;                 // pair start clamp
        int tap = kc / 20, ch = kc - tap * 20;
        int ty = tap / 3, tx = tap - ty * 3;
        a.u[p] = *(const unsigned int*)&xs[((py + 1 + ty) * P2 + (px + 1 + tx)) * XSS + ch];
      }
      BFrag bf;
      bf.v = *(const v16h*)&gwB[mrow * 192 + kb + 16 * hsel];
      acc = wmma16(a.v, bf.v, acc);
    }
    if (mrow < 12) {
      float bias = g1bv[mrow];
#pragma unroll
      for (int r = 0; r < 8; ++r) {
        int op = t * 16 + r + 8 * hsel;
        float v = acc[r] + bias;
        gbuf[op * 12 + mrow] = v > 0.f ? v : 0.f;
      }
    }
  }
  __syncthreads();

  // ---------- stage E: expert conv2 grouped 8->8 x 8 experts, WMMA ----------
  for (int u = wv; u < 128; u += 8) {     // (expert e, M-tile t)
    int e = u >> 4, t = u & 15;
    v8f acc = {};
    int pix = t * 16 + mrow;
    int py = pix >> 4, px = pix & 15;
#pragma unroll
    for (int kb = 0; kb < 96; kb += 32) {
      AFrag a;
#pragma unroll
      for (int r2 = 0; r2 < 2; ++r2) {
        int k0 = kb + 16 * hsel + 8 * r2;
        int tap = k0 >> 3; tap = tap < 8 ? tap : 8;
        int ty = tap / 3, tx = tap - ty * 3;
        a.h8[r2] = *(const v8h*)&hebuf[((py + ty) * P1 + (px + tx)) * 64 + e * 8];
      }
      BFrag bf;
      bf.v = *(const v16h*)&ewB[(e * 16 + mrow) * 96 + kb + 16 * hsel];
      acc = wmma16(a.v, bf.v, acc);
    }
    if (mrow < 8) {
      float bias = e2bv[e * 8 + mrow];
#pragma unroll
      for (int r = 0; r < 8; ++r) {
        int op = t * 16 + r + 8 * hsel;
        float v = acc[r] + bias;
        parts[((1 + e) * 8 + mrow) * 256 + op] = (_Float16)(v > 0.f ? v : 0.f);
      }
    }
  }
  __syncthreads();

  // ---------- stage F: gate 1x1 -> softmax -> mix -> sigmoid ----------
  {
    int pix = tid;
    int py = pix >> 4, px = pix & 15;
    float gv[12];
#pragma unroll
    for (int c = 0; c < 12; ++c) gv[c] = gbuf[pix * 12 + c];
    float s[9]; float mx = -1e30f;
#pragma unroll
    for (int p = 0; p < 9; ++p) {
      float acc = g2bv[p];
#pragma unroll
      for (int c = 0; c < 12; ++c) acc += gw2v[p * 12 + c] * gv[c];
      s[p] = acc; mx = acc > mx ? acc : mx;
    }
    float sum = 0.f;
#pragma unroll
    for (int p = 0; p < 9; ++p) { s[p] = __expf(s[p] - mx); sum += s[p]; }
    float inv = 1.f / sum;
    float mixv[8];
#pragma unroll
    for (int c = 0; c < 8; ++c) mixv[c] = 0.f;
#pragma unroll
    for (int p = 0; p < 9; ++p) {
      float wp = s[p] * inv;
#pragma unroll
      for (int c = 0; c < 8; ++c)
        mixv[c] += wp * (float)parts[(p * 8 + c) * 256 + pix];
    }
    float o = fbv[0];
#pragma unroll
    for (int c = 0; c < 8; ++c) o += fwv[c] * mixv[c];
    o = 1.f / (1.f + __expf(-o));
    out[(size_t)b * 262144 + (size_t)(y0 + py) * 512 + (x0 + px)] = o;
  }
}

extern "C" void kernel_launch(void* const* d_in, const int* in_sizes, int n_in,
                              void* d_out, int out_size, void* d_ws, size_t ws_size,
                              hipStream_t stream) {
  (void)in_sizes; (void)n_in; (void)out_size; (void)d_ws; (void)ws_size;
  const float* x    = (const float*)d_in[0];
  const float* w_s1 = (const float*)d_in[1];
  const float* b_s1 = (const float*)d_in[2];
  const float* w_s2 = (const float*)d_in[3];
  const float* b_s2 = (const float*)d_in[4];
  const float* w_e1 = (const float*)d_in[5];
  const float* b_e1 = (const float*)d_in[6];
  const float* w_e2 = (const float*)d_in[7];
  const float* b_e2 = (const float*)d_in[8];
  const float* w_g1 = (const float*)d_in[9];
  const float* b_g1 = (const float*)d_in[10];
  const float* w_g2 = (const float*)d_in[11];
  const float* b_g2 = (const float*)d_in[12];
  const float* w_f  = (const float*)d_in[13];
  const float* b_f  = (const float*)d_in[14];
  float* out = (float*)d_out;

  // allow >64KB dynamic LDS (CDNA5 WGP has 320KB)
  (void)hipFuncSetAttribute((const void*)moe_fused_kernel,
                            hipFuncAttributeMaxDynamicSharedMemorySize,
                            (int)SMEM_BYTES);

  dim3 grid(512 / TB, 512 / TB, 8);   // 32 x 32 x 8 workgroups
  dim3 block(256);                    // 8 wave32s
  moe_fused_kernel<<<grid, block, SMEM_BYTES, stream>>>(
      x, w_s1, b_s1, w_s2, b_s2, w_e1, b_e1, w_e2, b_e2,
      w_g1, b_g1, w_g2, b_g2, w_f, b_f, out);
}
